// Graph_40681930227717
// MI455X (gfx1250) — compile-verified, matrix operand order checked by reading, not run on hardware
//
#include <hip/hip_runtime.h>
#include <hip/hip_bf16.h>
#include <math.h>

// ---------------------------------------------------------------------------
// Types for CDNA5 WMMA (wave32): 16x16x32 bf16 -> f32
// ---------------------------------------------------------------------------
typedef __attribute__((ext_vector_type(16))) __bf16 bf16x16;
typedef __attribute__((ext_vector_type(8)))  __bf16 bf16x8;
typedef __attribute__((ext_vector_type(8)))  float  floatx8;

// ---------------------------------------------------------------------------
// WMMA GEMM:  C[M, Nout] (+)= A[M, K] * W[Nout, K]^T  (+bias)(relu)
// A, W are bf16 with K % 32 == 0, M % 16 == 0, Nout % 16 == 0.
// One wave computes one 16x16 tile; 8 waves/block cover a 128-wide N band.
// grid = (M/16, ceil(Nout/128)).
//
// Per-lane fragment layout (wave32, 16-bit A 16x32 / B 32x16):
//   lane = h*16 + r16 ; element e: K = (e<8 ? e : 8+e) + 8*h  (two contiguous
//   8-element chunks at k0+8h and k0+16+8h) -> two 16B vector loads each.
// ---------------------------------------------------------------------------
template <bool BIAS, bool ACCUM, bool RELU>
__global__ __launch_bounds__(256)
void wmma_gemm_kernel(const __bf16* __restrict__ A,
                      const __bf16* __restrict__ W,
                      const float* __restrict__ bias,
                      float* __restrict__ C,
                      int M, int K, int Nout)
{
    const int lane = threadIdx.x & 31;
    const int wave = threadIdx.x >> 5;
    const int h    = lane >> 4;
    const int r16  = lane & 15;
    const int m0   = blockIdx.x * 16;
    const int n0   = blockIdx.y * 128 + wave * 16;
    if (n0 >= Nout) return;

    const __bf16* Ap = A + (size_t)(m0 + r16) * K + 8 * h;
    const __bf16* Wp = W + (size_t)(n0 + r16) * K + 8 * h;

    floatx8 c = {};
    for (int k0 = 0; k0 < K; k0 += 32) {
        const bf16x8 a0 = *(const bf16x8*)(Ap + k0);
        const bf16x8 a1 = *(const bf16x8*)(Ap + k0 + 16);
        const bf16x8 b0 = *(const bf16x8*)(Wp + k0);
        const bf16x8 b1 = *(const bf16x8*)(Wp + k0 + 16);
        const bf16x16 a = __builtin_shufflevector(
            a0, a1, 0, 1, 2, 3, 4, 5, 6, 7, 8, 9, 10, 11, 12, 13, 14, 15);
        const bf16x16 b = __builtin_shufflevector(
            b0, b1, 0, 1, 2, 3, 4, 5, 6, 7, 8, 9, 10, 11, 12, 13, 14, 15);
        c = __builtin_amdgcn_wmma_f32_16x16x32_bf16(
                /*neg_a=*/false, a, /*neg_b=*/false, b,
                /*c_mod=*/(short)0, c, /*reuse_a=*/false, /*reuse_b=*/false);
    }

#pragma unroll
    for (int r = 0; r < 8; ++r) {
        const int row = m0 + r + 8 * h;
        const int col = n0 + r16;
        const size_t idx = (size_t)row * Nout + col;
        float v = c[r];
        if (BIAS)  v += bias[col];
        if (ACCUM) v += C[idx];
        if (RELU)  v = v > 0.0f ? v : 0.0f;
        C[idx] = v;
    }
}

// ---------------------------------------------------------------------------
// Elementwise f32 -> bf16 conversion
// ---------------------------------------------------------------------------
__global__ void cvt_bf16_kernel(const float* __restrict__ src,
                                __bf16* __restrict__ dst, long n)
{
    long t = (long)blockIdx.x * blockDim.x + threadIdx.x;
    if (t < n) dst[t] = (__bf16)src[t];
}

// f32 [rows, K] -> bf16 [rows, Kp] zero-padded in K
__global__ void pad_cvt_kernel(const float* __restrict__ src,
                               __bf16* __restrict__ dst,
                               long rows, int K, int Kp)
{
    long t = (long)blockIdx.x * blockDim.x + threadIdx.x;
    if (t >= rows * (long)Kp) return;
    const long r = t / Kp;
    const int  k = (int)(t - r * Kp);
    dst[t] = (k < K) ? (__bf16)src[r * K + k] : (__bf16)0.0f;
}

// ---------------------------------------------------------------------------
// B-spline basis (GRID=5, ORDER=3 -> 8 bases), knots g[t] = 0.4*(t-3) - 1.
// One thread per (row, feature); writes bf16 [rows, feat, 8].
// ---------------------------------------------------------------------------
__global__ void bspline_kernel(const float* __restrict__ X,
                               __bf16* __restrict__ out, long total)
{
    long t = (long)blockIdx.x * blockDim.x + threadIdx.x;
    if (t >= total) return;
    const float x = X[t];
    float b[11];
#pragma unroll
    for (int i = 0; i < 11; ++i) {
        const float g0 = 0.4f * (i - 3) - 1.0f;
        const float g1 = 0.4f * (i - 2) - 1.0f;
        b[i] = (x >= g0 && x < g1) ? 1.0f : 0.0f;
    }
#pragma unroll
    for (int k = 1; k <= 3; ++k) {
#pragma unroll
        for (int i = 0; i < 10; ++i) {
            if (i < 11 - k) {
                const float gi   = 0.4f * (i - 3) - 1.0f;
                const float gi1  = 0.4f * (i - 2) - 1.0f;
                const float gik  = 0.4f * (i - 3 + k) - 1.0f;
                const float gik1 = 0.4f * (i - 2 + k) - 1.0f;
                b[i] = (x - gi) / (gik - gi) * b[i]
                     + (gik1 - x) / (gik1 - gi1) * b[i + 1];
            }
        }
    }
    const size_t o = (size_t)t * 8;
#pragma unroll
    for (int s = 0; s < 8; ++s) out[o + s] = (__bf16)b[s];
}

// ---------------------------------------------------------------------------
// Misc small kernels
// ---------------------------------------------------------------------------
__global__ void copy_kernel(const float* __restrict__ src,
                            float* __restrict__ dst, long n)
{
    long t = (long)blockIdx.x * blockDim.x + threadIdx.x;
    if (t < n) dst[t] = src[t];
}

__global__ void edge_init_kernel(const int* __restrict__ ei,
                                 int* __restrict__ src, int* __restrict__ dst,
                                 int* __restrict__ emask, int E)
{
    int t = blockIdx.x * blockDim.x + threadIdx.x;
    if (t < E) { src[t] = ei[t]; dst[t] = ei[E + t]; emask[t] = 1; }
}

// GIN scatter: agg[dst] += x[src] for unmasked edges. 32 threads/edge, 4 f32 each.
__global__ void edge_scatter_kernel(const float* __restrict__ x,
                                    float* __restrict__ agg,
                                    const int* __restrict__ src,
                                    const int* __restrict__ dst,
                                    const int* __restrict__ emask, int E)
{
    long t = (long)blockIdx.x * blockDim.x + threadIdx.x;
    int e = (int)(t >> 5);
    int c = (int)(t & 31);
    if (e >= E || !emask[e]) return;
    const int s = src[e], d = dst[e];
    const float* xs = x + (size_t)s * 128 + c * 4;
    float* ad = agg + (size_t)d * 128 + c * 4;
#pragma unroll
    for (int q = 0; q < 4; ++q) atomicAdd(ad + q, xs[q]);
}

// score = sigmoid(x . p) ; one block (128 thr) per node, LDS reduce.
__global__ void score_kernel(const float* __restrict__ x,
                             const float* __restrict__ p,
                             float* __restrict__ score)
{
    __shared__ float sbuf[128];
    const int nidx = blockIdx.x;
    const int tid  = threadIdx.x;
    sbuf[tid] = x[(size_t)nidx * 128 + tid] * p[tid];
    __syncthreads();
    for (int s = 64; s > 0; s >>= 1) {
        if (tid < s) sbuf[tid] += sbuf[tid + s];
        __syncthreads();
    }
    if (tid == 0) score[nidx] = 1.0f / (1.0f + expf(-sbuf[0]));
}

// Per-graph top-k by rank (stable tie-break on index), gated gather + remap tables.
__global__ void topk_kernel(const float* __restrict__ xin,
                            const float* __restrict__ score,
                            float* __restrict__ xout,
                            int* __restrict__ newidx, int* __restrict__ keep,
                            int npg, int k)
{
    __shared__ float ss[256];
    __shared__ int   sf[256];
    const int g = blockIdx.x;
    const int tid = threadIdx.x;
    const int gnode = g * npg + tid;
    float my = 0.0f;
    if (tid < npg) { my = score[gnode]; ss[tid] = my; }
    __syncthreads();
    int selected = 0;
    if (tid < npg) {
        int rank = 0;
        for (int j = 0; j < npg; ++j) {
            const float sj = ss[j];
            rank += (sj > my) || (sj == my && j < tid);
        }
        selected = (rank < k) ? 1 : 0;
    }
    sf[tid] = selected;
    __syncthreads();
    if (tid < npg) {
        if (selected) {
            int pos = 0;
            for (int j = 0; j < tid; ++j) pos += sf[j];
            const int np = g * k + pos;
            newidx[gnode] = np;
            keep[gnode]   = 1;
            const float* xi = xin + (size_t)gnode * 128;
            float* xo = xout + (size_t)np * 128;
            for (int f = 0; f < 128; ++f) xo[f] = xi[f] * my;
        } else {
            newidx[gnode] = 0;
            keep[gnode]   = 0;
        }
    }
}

__global__ void edge_remap_kernel(int* __restrict__ src, int* __restrict__ dst,
                                  int* __restrict__ emask,
                                  const int* __restrict__ newidx,
                                  const int* __restrict__ keep, int E)
{
    int t = blockIdx.x * blockDim.x + threadIdx.x;
    if (t >= E) return;
    const int s = src[t], d = dst[t];
    emask[t] = emask[t] & keep[s] & keep[d];
    src[t] = newidx[s];
    dst[t] = newidx[d];
}

// readout: per graph [gmp | gap] over k packed nodes -> [G, 256]
__global__ void readout_kernel(const float* __restrict__ x,
                               float* __restrict__ out, int k)
{
    const int g = blockIdx.x, f = threadIdx.x; // 128 threads
    const float* base = x + (size_t)g * k * 128 + f;
    float mx = -INFINITY, sm = 0.0f;
    for (int i = 0; i < k; ++i) {
        const float v = base[(size_t)i * 128];
        mx = fmaxf(mx, v);
        sm += v;
    }
    out[(size_t)g * 256 + f]       = mx;
    out[(size_t)g * 256 + 128 + f] = sm / (float)k;
}

// readout fusion: branch0 -> relu(r0)+relu(r1); branch1 -> w0*relu(r0)+w1*relu(r1)+w2*relu(r2)
__global__ void combine_kernel(const float* __restrict__ r0,
                               const float* __restrict__ r1,
                               const float* __restrict__ r2,
                               const float* __restrict__ wr, int usew,
                               float* __restrict__ out, int n)
{
    int t = blockIdx.x * blockDim.x + threadIdx.x;
    if (t >= n) return;
    const float a = fmaxf(r0[t], 0.0f);
    const float b = fmaxf(r1[t], 0.0f);
    if (usew) {
        const float c = fmaxf(r2[t], 0.0f);
        out[t] = wr[0] * a + wr[1] * b + wr[2] * c;
    } else {
        out[t] = a + b;
    }
}

// fc_final (Nout=1): z[row] = xg[row,:].bw + bspl(xg)[row,:].sw_flat
__global__ void final_kernel(const float* __restrict__ xg,
                             const __bf16* __restrict__ bspl,
                             const float* __restrict__ bw,   // [256]
                             const float* __restrict__ sw,   // [2048]
                             float* __restrict__ z, int rows)
{
    int row = blockIdx.x * blockDim.x + threadIdx.x;
    if (row >= rows) return;
    float acc = 0.0f;
    const float* xr = xg + (size_t)row * 256;
    for (int i = 0; i < 256; ++i) acc += xr[i] * bw[i];
    const __bf16* br = bspl + (size_t)row * 2048;
    for (int i = 0; i < 2048; ++i) acc += (float)br[i] * sw[i];
    z[row] = acc;
}

// ---------------------------------------------------------------------------
// Orchestration
// ---------------------------------------------------------------------------
static inline long cdiv(long a, long b) { return (a + b - 1) / b; }

extern "C" void kernel_launch(void* const* d_in, const int* in_sizes, int n_in,
                              void* d_out, int out_size, void* d_ws, size_t ws_size,
                              hipStream_t stream)
{
    (void)n_in; (void)out_size; (void)ws_size;
    constexpr int G = 128, N = 256, NF = 128;
    const int NTOT = G * N;                 // 32768
    const int E = in_sizes[14] / 2;         // 524288

    // ---- workspace carve (256B aligned chunks) ----
    char* ws = (char*)d_ws;
    auto carve = [&](size_t bytes) -> void* {
        void* p = (void*)ws;
        ws += (bytes + 255) & ~(size_t)255;
        return p;
    };
    float*  bufA   = (float*) carve((size_t)NTOT * NF * 4);       // node feats (x)
    float*  bufB   = (float*) carve((size_t)NTOT * NF * 4);       // aggregation (f32, atomics)
    float*  bufC   = (float*) carve((size_t)NTOT * NF * 4);       // conv output
    __bf16* aggbf  = (__bf16*)carve((size_t)NTOT * NF * 2);       // agg in bf16
    __bf16* bspl   = (__bf16*)carve((size_t)NTOT * NF * 8 * 2);   // spline feats (bf16)
    __bf16* featp  = (__bf16*)carve((size_t)NTOT * 96 * 2);       // padded input feats
    __bf16* w0p    = (__bf16*)carve((size_t)NF * 96 * 2);         // padded input weight
    __bf16* cbwbf  = (__bf16*)carve((size_t)6 * NF * NF * 2);     // conv base weights
    __bf16* cswbf  = (__bf16*)carve((size_t)6 * NF * NF * 8 * 2); // conv spline weights
    __bf16* gbwbf  = (__bf16*)carve((size_t)2 * 256 * 256 * 2);   // fc_g base weights
    __bf16* gswbf  = (__bf16*)carve((size_t)2 * 256 * 2048 * 2);  // fc_g spline weights
    __bf16* xinbf  = (__bf16*)carve((size_t)G * 256 * 2);         // fused readout bf16
    float*  score  = (float*) carve((size_t)NTOT * 4);
    int*    nidx   = (int*)   carve((size_t)NTOT * 4);
    int*    keep   = (int*)   carve((size_t)NTOT * 4);
    int*    esrc   = (int*)   carve((size_t)E * 4);
    int*    edst   = (int*)   carve((size_t)E * 4);
    int*    emsk   = (int*)   carve((size_t)E * 4);
    float*  reads  = (float*) carve((size_t)3 * G * 256 * 4);     // 3 x [128,256]
    float*  xin    = (float*) carve((size_t)G * 256 * 4);         // [128,256]

    float* dout = (float*)d_out;

    // ---- one-time weight conversions (f32 -> bf16) ----
    cvt_bf16_kernel<<<cdiv(6L * NF * NF, 256), 256, 0, stream>>>(
        (const float*)d_in[6], cbwbf, 6L * NF * NF);
    cvt_bf16_kernel<<<cdiv(6L * NF * NF * 8, 256), 256, 0, stream>>>(
        (const float*)d_in[7], cswbf, 6L * NF * NF * 8);
    cvt_bf16_kernel<<<cdiv(2L * 256 * 256, 256), 256, 0, stream>>>(
        (const float*)d_in[8], gbwbf, 2L * 256 * 256);
    cvt_bf16_kernel<<<cdiv(2L * 256 * 2048, 256), 256, 0, stream>>>(
        (const float*)d_in[9], gswbf, 2L * 256 * 2048);

    for (int b = 0; b < 2; ++b) {
        const float* feat   = (const float*)d_in[b ? 1 : 0];
        const int    Fin    = b ? 43 : 93;
        const int    Finp   = b ? 64 : 96;   // padded to multiple of 32
        const float* W0     = (const float*)d_in[b ? 4 : 2];
        const float* bias0  = (const float*)d_in[b ? 5 : 3];
        const __bf16* convbw = cbwbf + (size_t)b * 3 * NF * NF;
        const __bf16* convsw = cswbf + (size_t)b * 3 * NF * NF * 8;
        const __bf16* fcgbw  = gbwbf + (size_t)b * 256 * 256;
        const __bf16* fcgsw  = gswbf + (size_t)b * 256 * 2048;
        const float* finbw  = (const float*)d_in[10] + (size_t)b * 256;
        const float* finsw  = (const float*)d_in[11] + (size_t)b * 2048;
        const float* poolp  = (const float*)d_in[12] + (size_t)b * 3 * NF;
        const float* wr     = (const float*)d_in[13];
        const int*   eidx   = (const int*)d_in[b ? 15 : 14];

        float* xg_out = dout + 128 + (size_t)b * G * 256;                 // x_g_all / x_g1
        float* z_out  = b ? (dout + 128 + 2 * (size_t)G * 256) : dout;    // z / z1

        // edges working copies
        edge_init_kernel<<<cdiv(E, 256), 256, 0, stream>>>(eidx, esrc, edst, emsk, E);

        // pad+convert input features / weight, then x0 = feat @ W0^T + bias0
        pad_cvt_kernel<<<cdiv((long)NTOT * Finp, 256), 256, 0, stream>>>(
            feat, featp, NTOT, Fin, Finp);
        pad_cvt_kernel<<<cdiv((long)NF * Finp, 256), 256, 0, stream>>>(
            W0, w0p, NF, Fin, Finp);
        wmma_gemm_kernel<true, false, false>
            <<<dim3(NTOT / 16, 1), 256, 0, stream>>>(featp, w0p, bias0, bufA, NTOT, Finp, NF);

        int npg = N;
        for (int lvl = 0; lvl < 3; ++lvl) {
            const int ntot = G * npg;
            const long nf  = (long)ntot * NF;

            // agg = x + segment_sum(x[src] -> dst)   (f32 atomics)
            copy_kernel<<<cdiv(nf, 256), 256, 0, stream>>>(bufA, bufB, nf);
            edge_scatter_kernel<<<cdiv((long)E * 32, 256), 256, 0, stream>>>(
                bufA, bufB, esrc, edst, emsk, E);

            // bf16 copy of agg + spline features of agg
            cvt_bf16_kernel<<<cdiv(nf, 256), 256, 0, stream>>>(bufB, aggbf, nf);
            bspline_kernel<<<cdiv(nf, 256), 256, 0, stream>>>(bufB, bspl, nf);

            // KAN: base GEMM then spline GEMM (accumulate) + relu
            wmma_gemm_kernel<false, false, false>
                <<<dim3(ntot / 16, 1), 256, 0, stream>>>(
                    aggbf, convbw + (size_t)lvl * NF * NF, nullptr, bufC, ntot, NF, NF);
            wmma_gemm_kernel<false, true, true>
                <<<dim3(ntot / 16, 1), 256, 0, stream>>>(
                    bspl, convsw + (size_t)lvl * NF * NF * 8, nullptr, bufC, ntot, NF * 8, NF);

            // mvpool
            score_kernel<<<ntot, 128, 0, stream>>>(bufC, poolp + (size_t)lvl * NF, score);
            const int k = npg / 2;
            topk_kernel<<<G, 256, 0, stream>>>(bufC, score, bufA, nidx, keep, npg, k);
            edge_remap_kernel<<<cdiv(E, 256), 256, 0, stream>>>(esrc, edst, emsk, nidx, keep, E);

            // readout
            readout_kernel<<<G, 128, 0, stream>>>(bufA, reads + (size_t)lvl * G * 256, k);
            npg = k;
        }

        // fuse readouts
        combine_kernel<<<cdiv((long)G * 256, 256), 256, 0, stream>>>(
            reads, reads + (size_t)G * 256, reads + (size_t)2 * G * 256,
            wr, b, xin, G * 256);

        // fc_g KAN: [128,256] -> [128,256], written straight into d_out
        cvt_bf16_kernel<<<cdiv((long)G * 256, 256), 256, 0, stream>>>(xin, xinbf, (long)G * 256);
        bspline_kernel<<<cdiv((long)G * 256, 256), 256, 0, stream>>>(xin, bspl, (long)G * 256);
        wmma_gemm_kernel<false, false, false>
            <<<dim3(G / 16, 2), 256, 0, stream>>>(xinbf, fcgbw, nullptr, xg_out, G, 256, 256);
        wmma_gemm_kernel<false, true, false>
            <<<dim3(G / 16, 2), 256, 0, stream>>>(bspl, fcgsw, nullptr, xg_out, G, 2048, 256);

        // fc_final KAN (Nout = 1)
        bspline_kernel<<<cdiv((long)G * 256, 256), 256, 0, stream>>>(xg_out, bspl, (long)G * 256);
        final_kernel<<<1, 128, 0, stream>>>(xg_out, bspl, finbw, finsw, z_out, G);
    }
}
